// MultisourceConvAttention_3736621547617
// MI455X (gfx1250) — compile-verified, matrix operand order checked by reading, not run on hardware
//
#include <hip/hip_runtime.h>
#include <hip/hip_bf16.h>

// ---------------- types ----------------
typedef __attribute__((ext_vector_type(16))) __bf16 v16bf;
typedef __attribute__((ext_vector_type(8)))  __bf16 v8bf;
typedef __attribute__((ext_vector_type(8)))  float  v8f;

#define NKEY 2240
#define NQ   1600

__device__ __forceinline__ float gelu_f(float x) {
    float x3 = x * x * x;
    return 0.5f * x * (1.0f + tanhf(0.7978845608028654f * (x + 0.044715f * x3)));
}

// Load a 16x32 bf16 WMMA A/B fragment from row-major memory (ld elements per row).
// Lane L: row = L%16 ; K offsets (L/16)*8 + [0..7] and 16 + (L/16)*8 + [0..7].
__device__ __forceinline__ v16bf load_frag_bf(const __bf16* __restrict__ base, int ld, int lane) {
    int r    = lane & 15;
    int koff = (lane >> 4) * 8;
    const __bf16* p = base + (size_t)r * ld + koff;
    v8bf lo = *(const v8bf*)(p);
    v8bf hi = *(const v8bf*)(p + 16);
    v16bf f;
#pragma unroll
    for (int i = 0; i < 8; ++i) { f[i] = lo[i]; f[8 + i] = hi[i]; }
    return f;
}

// Same fragment but sourced from f32 (LDS) with on-the-fly conversion.
__device__ __forceinline__ v16bf frag_from_f32(const float* __restrict__ base, int ld, int lane) {
    int r    = lane & 15;
    int koff = (lane >> 4) * 8;
    const float* p = base + (size_t)r * ld + koff;
    v16bf f;
#pragma unroll
    for (int i = 0; i < 8; ++i) { f[i] = (__bf16)p[i]; f[8 + i] = (__bf16)p[16 + i]; }
    return f;
}

__device__ __forceinline__ v8f zero8() {
    v8f z = {0.f, 0.f, 0.f, 0.f, 0.f, 0.f, 0.f, 0.f};
    return z;
}

// ---------------- weight transpose + bf16 convert: Wt[n][k] = W[k][n] ----------------
__global__ void transpose_bf_kernel(const float* __restrict__ W, __bf16* __restrict__ Wt,
                                    int K, int N) {
    int idx = blockIdx.x * 256 + threadIdx.x;
    if (idx < K * N) {
        int k = idx / N, n = idx % N;
        Wt[(size_t)n * K + k] = (__bf16)W[(size_t)k * N + n];
    }
}

// ---------------- patchify keys -> bf16 (B,2240,1024) ----------------
__global__ __launch_bounds__(256) void patchify_k_kernel(const float* __restrict__ k0,
                                                         const float* __restrict__ k1,
                                                         const float* __restrict__ k2,
                                                         __bf16* __restrict__ out) {
    int row = blockIdx.x;                 // b*2240 + key
    int b = row / NKEY, key = row % NKEY;
    const float* src; int Himg, Wimg, wpat, local;
    if (key < 1024)      { src = k0; Himg = 256; Wimg = 256; wpat = 32; local = key; }
    else if (key < 1984) { src = k1; Himg = 192; Wimg = 320; wpat = 40; local = key - 1024; }
    else                 { src = k2; Himg = 128; Wimg = 128; wpat = 16; local = key - 1984; }
    int hy = local / wpat, wx = local % wpat;
    for (int e = threadIdx.x; e < 1024; e += 256) {
        int c = e & 15, pw = (e >> 4) & 7, ph = e >> 7;
        float v = src[(((size_t)b * 16 + c) * Himg + hy * 8 + ph) * Wimg + wx * 8 + pw];
        out[(size_t)row * 1024 + e] = (__bf16)v;
    }
}

// ---------------- patchify queries -> bf16 (B,1600,1024) ----------------
__global__ __launch_bounds__(256) void patchify_q_kernel(const float* __restrict__ q0,
                                                         const float* __restrict__ q1,
                                                         __bf16* __restrict__ out) {
    int row = blockIdx.x;                 // b*1600 + qidx
    int b = row / NQ, key = row % NQ;
    const float* src; int Himg, Wimg, wpat, local;
    if (key < 1024) { src = q0; Himg = 256; Wimg = 256; wpat = 32; local = key; }
    else            { src = q1; Himg = 192; Wimg = 192; wpat = 24; local = key - 1024; }
    int hy = local / wpat, wx = local % wpat;
    for (int e = threadIdx.x; e < 1024; e += 256) {
        int c = e & 15, pw = (e >> 4) & 7, ph = e >> 7;
        float v = src[(((size_t)b * 16 + c) * Himg + hy * 8 + ph) * Wimg + wx * 8 + pw];
        out[(size_t)row * 1024 + e] = (__bf16)v;
    }
}

// -------- values: patchify + layernorm(2048) + head-split transpose -> v_t[b][h][256][2240] bf16 --------
__global__ __launch_bounds__(256) void v_ln_kernel(const float* __restrict__ v0,
                                                   const float* __restrict__ v1,
                                                   const float* __restrict__ v2,
                                                   const float* __restrict__ g,
                                                   const float* __restrict__ beta,
                                                   __bf16* __restrict__ v_t) {
    __shared__ float rs[256], rq[256];
    int row = blockIdx.x;                 // b*2240 + key
    int b = row / NKEY, key = row % NKEY;
    const float* src; int Himg, Wimg, wpat, local;
    if (key < 1024)      { src = v0; Himg = 256; Wimg = 256; wpat = 32; local = key; }
    else if (key < 1984) { src = v1; Himg = 192; Wimg = 320; wpat = 40; local = key - 1024; }
    else                 { src = v2; Himg = 128; Wimg = 128; wpat = 16; local = key - 1984; }
    int hy = local / wpat, wx = local % wpat;
    int tid = threadIdx.x;
    float vals[8];
    float s = 0.f, sq = 0.f;
#pragma unroll
    for (int i = 0; i < 8; ++i) {
        int e = tid + i * 256;            // 0..2047
        int c = e & 31, pw = (e >> 5) & 7, ph = e >> 8;
        float v = src[(((size_t)b * 32 + c) * Himg + hy * 8 + ph) * Wimg + wx * 8 + pw];
        vals[i] = v; s += v; sq += v * v;
    }
    rs[tid] = s; rq[tid] = sq;
    __syncthreads();
    for (int st = 128; st > 0; st >>= 1) {
        if (tid < st) { rs[tid] += rs[tid + st]; rq[tid] += rq[tid + st]; }
        __syncthreads();
    }
    float mean = rs[0] * (1.0f / 2048.0f);
    float var  = rq[0] * (1.0f / 2048.0f) - mean * mean;
    float inv  = rsqrtf(var + 1e-5f);
#pragma unroll
    for (int i = 0; i < 8; ++i) {
        int e = tid + i * 256;
        float nv = (vals[i] - mean) * inv * g[e] + beta[e];
        int hd = e >> 8, dd = e & 255;
        v_t[(((size_t)b * 8 + hd) * 256 + dd) * NKEY + key] = (__bf16)nv;
    }
}

// ---------------- generic WMMA GEMM: out = act(A(MxK) @ Wt^T + bias), N = 256 ----------------
// Ping-pong software pipeline (2x unrolled, peeled epilogue): the two fragment buffer
// sets alternate roles structurally, so loads land directly in their final registers
// (no v_mov rotation) and the scheduler can overlap next-step loads with v_wmma.
// Requires K to be a multiple of 64 (steps even) -- true for all call sites.
// mode 0: out bf16 row-major (M,256)
// mode 1: head-split: out[((b*8+h)*seq + s)*32 + d], h = n/32, d = n%32
__global__ __launch_bounds__(128) void gemm_wmma_kernel(const __bf16* __restrict__ A,
                                                        const __bf16* __restrict__ Wt,
                                                        const float* __restrict__ bias,
                                                        __bf16* __restrict__ out,
                                                        int M, int K, int mode, int seq_len,
                                                        int do_gelu) {
    int lane = threadIdx.x & 31, wave = threadIdx.x >> 5;
    int m0 = blockIdx.x * 16;
    int n0 = wave * 64;
    const __bf16* Abase = A + (size_t)m0 * K;
    v8f acc[4];
#pragma unroll
    for (int j = 0; j < 4; ++j) acc[j] = zero8();

    int steps = K >> 5;                 // even, >= 2
    v16bf a0 = load_frag_bf(Abase, K, lane);
    v16bf b0[4];
#pragma unroll
    for (int j = 0; j < 4; ++j)
        b0[j] = load_frag_bf(Wt + (size_t)(n0 + j * 16) * K, K, lane);
    v16bf a1;
    v16bf b1[4];

    for (int s = 0; s < steps - 2; s += 2) {
        int k1 = (s + 1) << 5, k2 = (s + 2) << 5;
        a1 = load_frag_bf(Abase + k1, K, lane);
#pragma unroll
        for (int j = 0; j < 4; ++j)
            b1[j] = load_frag_bf(Wt + (size_t)(n0 + j * 16) * K + k1, K, lane);
#pragma unroll
        for (int j = 0; j < 4; ++j)
            acc[j] = __builtin_amdgcn_wmma_f32_16x16x32_bf16(false, a0, false, b0[j],
                                                             (short)0, acc[j], false, false);
        a0 = load_frag_bf(Abase + k2, K, lane);
#pragma unroll
        for (int j = 0; j < 4; ++j)
            b0[j] = load_frag_bf(Wt + (size_t)(n0 + j * 16) * K + k2, K, lane);
#pragma unroll
        for (int j = 0; j < 4; ++j)
            acc[j] = __builtin_amdgcn_wmma_f32_16x16x32_bf16(false, a1, false, b1[j],
                                                             (short)0, acc[j], false, false);
    }
    {   // final pair
        int k1 = (steps - 1) << 5;
        a1 = load_frag_bf(Abase + k1, K, lane);
#pragma unroll
        for (int j = 0; j < 4; ++j)
            b1[j] = load_frag_bf(Wt + (size_t)(n0 + j * 16) * K + k1, K, lane);
#pragma unroll
        for (int j = 0; j < 4; ++j)
            acc[j] = __builtin_amdgcn_wmma_f32_16x16x32_bf16(false, a0, false, b0[j],
                                                             (short)0, acc[j], false, false);
#pragma unroll
        for (int j = 0; j < 4; ++j)
            acc[j] = __builtin_amdgcn_wmma_f32_16x16x32_bf16(false, a1, false, b1[j],
                                                             (short)0, acc[j], false, false);
    }

    int mh = (lane >> 4) * 8, nn = lane & 15;
#pragma unroll
    for (int j = 0; j < 4; ++j) {
        int n = n0 + j * 16 + nn;
        float bn = bias[n];
#pragma unroll
        for (int v = 0; v < 8; ++v) {
            int row = m0 + mh + v;
            float val = acc[j][v] + bn;
            if (do_gelu) val = gelu_f(val);
            if (mode == 0) {
                out[(size_t)row * 256 + n] = (__bf16)val;
            } else {
                int bb = row / seq_len, sp = row % seq_len;
                int hh = n >> 5, d = n & 31;
                out[(((size_t)bb * 8 + hh) * seq_len + sp) * 32 + d] = (__bf16)val;
            }
        }
    }
}

// ---------------- fused attention: QK^T -> softmax -> segmented P@V -> scatter to mbconv input ----------------
__global__ __launch_bounds__(256) void attn_fused_kernel(const __bf16* __restrict__ q_emb,
                                                         const __bf16* __restrict__ k_emb,
                                                         const __bf16* __restrict__ v_t,
                                                         float* __restrict__ mb0,
                                                         float* __restrict__ mb1) {
    __shared__ float S[16][NKEY];    // 16 queries x 2240 keys, f32 -> later holds probabilities
    __shared__ float red[16][16];
    const int qt = blockIdx.x, h = blockIdx.y, b = blockIdx.z;
    const int tid = threadIdx.x, lane = tid & 31, wave = tid >> 5;
    const int bh = b * 8 + h;
    int mh = (lane >> 4) * 8, nn = lane & 15;

    // ---- phase A: S = (Q K^T) * scale; two independent K-tiles in flight per iter ----
    {
        const __bf16* qbase = q_emb + ((size_t)bh * NQ + qt * 16) * 32;
        const __bf16* kbase = k_emb + (size_t)bh * NKEY * 32;
        v16bf qa = load_frag_bf(qbase, 32, lane);
        const float scale = 0.17677669529663689f;   // 1/sqrt(32)
        int kt = wave;                              // 140 tiles, 8 waves
        for (; kt + 8 < NKEY / 16; kt += 16) {
            v16bf kb0 = load_frag_bf(kbase + (size_t)kt * 512, 32, lane);
            v16bf kb1 = load_frag_bf(kbase + (size_t)(kt + 8) * 512, 32, lane);
            v8f acc0 = zero8(), acc1 = zero8();
            acc0 = __builtin_amdgcn_wmma_f32_16x16x32_bf16(false, qa, false, kb0,
                                                           (short)0, acc0, false, false);
            acc1 = __builtin_amdgcn_wmma_f32_16x16x32_bf16(false, qa, false, kb1,
                                                           (short)0, acc1, false, false);
#pragma unroll
            for (int v = 0; v < 8; ++v) {
                S[mh + v][kt * 16 + nn]       = acc0[v] * scale;
                S[mh + v][(kt + 8) * 16 + nn] = acc1[v] * scale;
            }
        }
        if (kt < NKEY / 16) {                       // odd leftover tile (waves 4..7)
            v16bf kb0 = load_frag_bf(kbase + (size_t)kt * 512, 32, lane);
            v8f acc0 = zero8();
            acc0 = __builtin_amdgcn_wmma_f32_16x16x32_bf16(false, qa, false, kb0,
                                                           (short)0, acc0, false, false);
#pragma unroll
            for (int v = 0; v < 8; ++v)
                S[mh + v][kt * 16 + nn] = acc0[v] * scale;
        }
    }
    __syncthreads();

    // ---- softmax over 2240 keys per query row ----
    {
        int row = tid >> 4, sub = tid & 15;
        float mx = -3.0e38f;
        for (int j = sub; j < NKEY; j += 16) mx = fmaxf(mx, S[row][j]);
        red[row][sub] = mx;
        __syncthreads();
        mx = red[row][0];
#pragma unroll
        for (int i = 1; i < 16; ++i) mx = fmaxf(mx, red[row][i]);
        __syncthreads();
        float sum = 0.f;
        for (int j = sub; j < NKEY; j += 16) {
            float e = __expf(S[row][j] - mx);
            S[row][j] = e;
            sum += e;
        }
        red[row][sub] = sum;
        __syncthreads();
        sum = 0.f;
#pragma unroll
        for (int i = 0; i < 16; ++i) sum += red[row][i];
        float inv = 1.0f / sum;
        for (int j = sub; j < NKEY; j += 16) S[row][j] *= inv;
    }
    __syncthreads();

    // ---- phase C: per-segment O = P @ V, ping-pong pipelined, scatter to mbconv inputs ----
    const int seg_off[4] = {0, 1024, 1984, 2240};
    const float* pbase = &S[0][0];
    for (int p = wave; p < 48; p += 8) {
        int seg = p % 3, dt = p / 3;   // 3 segments x 16 d-tiles (of 256 value dims)
        v8f acc = zero8();
        const __bf16* vb = v_t + ((size_t)bh * 256 + dt * 16) * NKEY;
        int start = seg_off[seg];
        int steps = (seg_off[seg + 1] - start) >> 5;   // 32 / 30 / 8 : even
        v16bf pa0 = frag_from_f32(pbase + start, NKEY, lane);
        v16bf vf0 = load_frag_bf(vb + start, NKEY, lane);
        v16bf pa1, vf1;
        for (int i = 0; i < steps - 2; i += 2) {
            int k1 = start + ((i + 1) << 5), k2 = start + ((i + 2) << 5);
            pa1 = frag_from_f32(pbase + k1, NKEY, lane);
            vf1 = load_frag_bf(vb + k1, NKEY, lane);
            acc = __builtin_amdgcn_wmma_f32_16x16x32_bf16(false, pa0, false, vf0,
                                                          (short)0, acc, false, false);
            pa0 = frag_from_f32(pbase + k2, NKEY, lane);
            vf0 = load_frag_bf(vb + k2, NKEY, lane);
            acc = __builtin_amdgcn_wmma_f32_16x16x32_bf16(false, pa1, false, vf1,
                                                          (short)0, acc, false, false);
        }
        int klast = start + ((steps - 1) << 5);
        pa1 = frag_from_f32(pbase + klast, NKEY, lane);
        vf1 = load_frag_bf(vb + klast, NKEY, lane);
        acc = __builtin_amdgcn_wmma_f32_16x16x32_bf16(false, pa0, false, vf0,
                                                      (short)0, acc, false, false);
        acc = __builtin_amdgcn_wmma_f32_16x16x32_bf16(false, pa1, false, vf1,
                                                      (short)0, acc, false, false);

        int d  = dt * 16 + nn;
        int ph = d >> 5, pw = (d >> 2) & 7, cc = d & 3;
        int ch = h * 12 + seg * 4 + cc;
#pragma unroll
        for (int v = 0; v < 8; ++v) {
            int qg = qt * 16 + mh + v;
            float val = acc[v];
            if (qg < 1024) {            // query image 0: 32x32 patches, 256x256
                int hy = qg >> 5, wx = qg & 31;
                mb0[(((size_t)b * 96 + ch) * 256 + hy * 8 + ph) * 256 + wx * 8 + pw] = val;
            } else {                    // query image 1: 24x24 patches, 192x192
                int lq = qg - 1024;
                int hy = lq / 24, wx = lq % 24;
                mb1[(((size_t)b * 96 + ch) * 192 + hy * 8 + ph) * 192 + wx * 8 + pw] = val;
            }
        }
    }
}

// ---------------- fused MBConv: 1x1 grouped expand + gelu -> dw7x7 + gelu -> 1x1 grouped proj ----------------
__global__ __launch_bounds__(256) void mbconv_kernel(const float* __restrict__ in,   // (B,96,H,W)
                                                     float* __restrict__ out,        // (B,32,H,W)
                                                     const float* __restrict__ w_exp,
                                                     const float* __restrict__ b_exp,
                                                     const float* __restrict__ w_dw,
                                                     const float* __restrict__ b_dw,
                                                     const float* __restrict__ w_proj,
                                                     const float* __restrict__ b_proj,
                                                     int H, int W) {
    __shared__ float inT[12][22 * 22];
    __shared__ float hid[48][22 * 22];
    int tilesX = W >> 4;
    int tile = blockIdx.x, g = blockIdx.y, b = blockIdx.z;
    int ty0 = (tile / tilesX) * 16, tx0 = (tile % tilesX) * 16;
    int tid = threadIdx.x;

    for (int idx = tid; idx < 12 * 484; idx += 256) {
        int ic = idx / 484, pos = idx % 484;
        int y = pos / 22, x = pos % 22;
        int gy = ty0 + y - 3, gx = tx0 + x - 3;
        float v = 0.f;
        if (gy >= 0 && gy < H && gx >= 0 && gx < W)
            v = in[(((size_t)b * 96 + g * 12 + ic) * H + gy) * W + gx];
        inT[ic][pos] = v;
    }
    __syncthreads();

    for (int idx = tid; idx < 48 * 484; idx += 256) {
        int hc = idx / 484, pos = idx % 484;
        int y = pos / 22, x = pos % 22;
        int gy = ty0 + y - 3, gx = tx0 + x - 3;
        float v = 0.f;
        if (gy >= 0 && gy < H && gx >= 0 && gx < W) {
            float s = b_exp[g * 48 + hc];
#pragma unroll
            for (int ic = 0; ic < 12; ++ic)
                s += w_exp[(g * 48 + hc) * 12 + ic] * inT[ic][pos];
            v = gelu_f(s);
        }
        hid[hc][pos] = v;
    }
    __syncthreads();

    // one pixel per thread; 4 output channels accumulated together
    int pos = tid;
    int y = pos >> 4, x = pos & 15;
    float a0 = b_proj[g * 4 + 0], a1 = b_proj[g * 4 + 1];
    float a2 = b_proj[g * 4 + 2], a3 = b_proj[g * 4 + 3];
    for (int hc = 0; hc < 48; ++hc) {
        float ds = b_dw[g * 48 + hc];
        const float* wd = w_dw + (size_t)(g * 48 + hc) * 49;
#pragma unroll
        for (int ky = 0; ky < 7; ++ky)
#pragma unroll
            for (int kx = 0; kx < 7; ++kx)
                ds += wd[ky * 7 + kx] * hid[hc][(y + ky) * 22 + (x + kx)];
        float gd = gelu_f(ds);
        const float* wp = w_proj + hc;
        a0 += wp[(size_t)(g * 4 + 0) * 48] * gd;
        a1 += wp[(size_t)(g * 4 + 1) * 48] * gd;
        a2 += wp[(size_t)(g * 4 + 2) * 48] * gd;
        a3 += wp[(size_t)(g * 4 + 3) * 48] * gd;
    }
    size_t obase = (((size_t)b * 32 + g * 4) * H + ty0 + y) * W + tx0 + x;
    size_t cstride = (size_t)H * W;
    out[obase + 0 * cstride] = a0;
    out[obase + 1 * cstride] = a1;
    out[obase + 2 * cstride] = a2;
    out[obase + 3 * cstride] = a3;
}

// ---------------- host launcher ----------------
extern "C" void kernel_launch(void* const* d_in, const int* in_sizes, int n_in,
                              void* d_out, int out_size, void* d_ws, size_t ws_size,
                              hipStream_t stream) {
    (void)in_sizes; (void)n_in; (void)out_size; (void)ws_size;
    const float* keys0  = (const float*)d_in[0];
    const float* keys1  = (const float*)d_in[1];
    const float* keys2  = (const float*)d_in[2];
    const float* vals0  = (const float*)d_in[3];
    const float* vals1  = (const float*)d_in[4];
    const float* vals2  = (const float*)d_in[5];
    const float* quer0  = (const float*)d_in[6];
    const float* quer1  = (const float*)d_in[7];
    const float* ln_g   = (const float*)d_in[8];
    const float* ln_b   = (const float*)d_in[9];
    const float* kW1    = (const float*)d_in[10];
    const float* kb1    = (const float*)d_in[11];
    const float* kW2    = (const float*)d_in[12];
    const float* kb2    = (const float*)d_in[13];
    const float* qW1    = (const float*)d_in[14];
    const float* qb1    = (const float*)d_in[15];
    const float* qW2    = (const float*)d_in[16];
    const float* qb2    = (const float*)d_in[17];
    const float* w_exp  = (const float*)d_in[18];
    const float* b_exp  = (const float*)d_in[19];
    const float* w_dw   = (const float*)d_in[20];
    const float* b_dw   = (const float*)d_in[21];
    const float* w_proj = (const float*)d_in[22];
    const float* b_proj = (const float*)d_in[23];

    size_t off = 0;
    auto alloc = [&](size_t bytes) -> void* {
        void* p = (char*)d_ws + off;
        off += (bytes + 255) & ~(size_t)255;
        return p;
    };
    __bf16* k_seq = (__bf16*)alloc((size_t)2 * NKEY * 1024 * 2);
    __bf16* q_seq = (__bf16*)alloc((size_t)2 * NQ * 1024 * 2);
    __bf16* v_t   = (__bf16*)alloc((size_t)2 * 8 * 256 * NKEY * 2);
    __bf16* kW1t  = (__bf16*)alloc((size_t)1024 * 256 * 2);
    __bf16* kW2t  = (__bf16*)alloc((size_t)256 * 256 * 2);
    __bf16* qW1t  = (__bf16*)alloc((size_t)1024 * 256 * 2);
    __bf16* qW2t  = (__bf16*)alloc((size_t)256 * 256 * 2);
    __bf16* k_hid = (__bf16*)alloc((size_t)2 * NKEY * 256 * 2);
    __bf16* q_hid = (__bf16*)alloc((size_t)2 * NQ * 256 * 2);
    __bf16* k_emb = (__bf16*)alloc((size_t)2 * 8 * NKEY * 32 * 2);
    __bf16* q_emb = (__bf16*)alloc((size_t)2 * 8 * NQ * 32 * 2);
    float*  mb0   = (float*)alloc((size_t)2 * 96 * 256 * 256 * 4);
    float*  mb1   = (float*)alloc((size_t)2 * 96 * 192 * 192 * 4);

    float* out0 = (float*)d_out;                               // (2,32,256,256)
    float* out1 = (float*)d_out + (size_t)2 * 32 * 256 * 256;  // (2,32,192,192)

    // weights -> bf16 transposed
    transpose_bf_kernel<<<1024, 256, 0, stream>>>(kW1, kW1t, 1024, 256);
    transpose_bf_kernel<<<256,  256, 0, stream>>>(kW2, kW2t, 256, 256);
    transpose_bf_kernel<<<1024, 256, 0, stream>>>(qW1, qW1t, 1024, 256);
    transpose_bf_kernel<<<256,  256, 0, stream>>>(qW2, qW2t, 256, 256);

    // patchify + layernorm
    patchify_k_kernel<<<2 * NKEY, 256, 0, stream>>>(keys0, keys1, keys2, k_seq);
    patchify_q_kernel<<<2 * NQ,   256, 0, stream>>>(quer0, quer1, q_seq);
    v_ln_kernel<<<2 * NKEY, 256, 0, stream>>>(vals0, vals1, vals2, ln_g, ln_b, v_t);

    // embeddings (WMMA GEMMs)
    gemm_wmma_kernel<<<(2 * NKEY) / 16, 128, 0, stream>>>(k_seq, kW1t, kb1, k_hid,
                                                          2 * NKEY, 1024, 0, 0, 1);
    gemm_wmma_kernel<<<(2 * NKEY) / 16, 128, 0, stream>>>(k_hid, kW2t, kb2, k_emb,
                                                          2 * NKEY, 256, 1, NKEY, 0);
    gemm_wmma_kernel<<<(2 * NQ) / 16, 128, 0, stream>>>(q_seq, qW1t, qb1, q_hid,
                                                        2 * NQ, 1024, 0, 0, 1);
    gemm_wmma_kernel<<<(2 * NQ) / 16, 128, 0, stream>>>(q_hid, qW2t, qb2, q_emb,
                                                        2 * NQ, 256, 1, NQ, 0);

    // fused attention -> mbconv inputs
    attn_fused_kernel<<<dim3(NQ / 16, 8, 2), 256, 0, stream>>>(q_emb, k_emb, v_t, mb0, mb1);

    // mbconv per query image
    mbconv_kernel<<<dim3(16 * 16, 8, 2), 256, 0, stream>>>(mb0, out0, w_exp, b_exp,
                                                           w_dw, b_dw, w_proj, b_proj, 256, 256);
    mbconv_kernel<<<dim3(12 * 12, 8, 2), 256, 0, stream>>>(mb1, out1, w_exp, b_exp,
                                                           w_dw, b_dw, w_proj, b_proj, 192, 192);
}
